// ONNXCompatibleDGCNNBackbone_9964324127010
// MI455X (gfx1250) — matrix operand order repaired and checked by reading, not compiled, not run on hardware
//
#include <hip/hip_runtime.h>

#define NPTS 10000
#define BATCH 2

typedef _Float16 h8   __attribute__((ext_vector_type(8)));
typedef _Float16 v16h __attribute__((ext_vector_type(16)));
typedef float    v8f  __attribute__((ext_vector_type(8)));

__device__ __forceinline__ v16h cat16(h8 a, h8 b) {
  return __builtin_shufflevector(a, b, 0,1,2,3,4,5,6,7,8,9,10,11,12,13,14,15);
}

// --------------------------- TDM staging (guarded) -------------------------
// ROCm 7.2 clang-22: 5-arg builtin, no TDM header.
// amdgpu-toolchain (clang-23 + therock headers): TDM header present, 6-arg
// builtin (uint32x4, int32x8, int32x4, int32x4, int32x8, i32).
#if defined(__has_builtin)
#  if __has_builtin(__builtin_amdgcn_tensor_load_to_lds) && \
      __has_builtin(__builtin_amdgcn_s_wait_tensorcnt)
#    if __has_include(<hip/amd_detail/amd_gfx1250_TDM.h>)
#      define USE_TDM 2   /* 6-arg toolchain */
#    else
#      define USE_TDM 1   /* 5-arg toolchain */
#    endif
#  endif
#endif
#ifndef USE_TDM
#define USE_TDM 0
#endif

#if USE_TDM
typedef unsigned int v4u __attribute__((ext_vector_type(4)));
typedef int          v8i __attribute__((ext_vector_type(8)));
typedef int          v4i __attribute__((ext_vector_type(4)));

// DMA a 2-D tile (nrows x width f16 elements, row pitch = stride elements)
// from global memory into LDS at byte offset lds_off.  D# layout per CDNA5
// ISA ch.8: group0 = {flags/count, lds_addr, global_addr, type=2},
// group1 = {data_size=2B, tensor_dim0/1, tile_dim0/1, dim0_stride}.
__device__ __forceinline__ void tdm_load_rows(
    unsigned lds_off, const _Float16* gbase, int nrows, int width, int stride)
{
  unsigned long long ga = (unsigned long long)(size_t)gbase;
  v4u g0;
  g0[0] = 1u;                                        // count=1 (valid user D#)
  g0[1] = lds_off;                                   // lds_addr (bytes)
  g0[2] = (unsigned)(ga & 0xffffffffu);              // global_addr[31:0]
  g0[3] = (unsigned)((ga >> 32) & 0x01ffffffu)       // global_addr[56:32]
        | (2u << 30);                                // type=2 ("image")
  v8i g1;
  g1[0] = 0x00010000;                                // data_size=1 (2 bytes)
  g1[1] = (int)((unsigned)width << 16);              // tensor_dim0[15:0]
  g1[2] = (int)(((unsigned)width >> 16) & 0xffffu)   // tensor_dim0[31:16]
        | (int)((unsigned)nrows << 16);              // tensor_dim1[15:0]
  g1[3] = (int)(((unsigned)nrows >> 16) & 0xffffu)   // tensor_dim1[31:16]
        | (int)((unsigned)width << 16);              // tile_dim0
  g1[4] = nrows & 0xffff;                            // tile_dim1 (tile_dim2=0)
  g1[5] = stride;                                    // tensor_dim0_stride lo32
  g1[6] = 0;
  g1[7] = 0;
  v4i z4 = {0, 0, 0, 0};
#if USE_TDM == 2
  v8i z8 = {0, 0, 0, 0, 0, 0, 0, 0};
  __builtin_amdgcn_tensor_load_to_lds(g0, g1, z4, z4, z8, 0);
#else
  __builtin_amdgcn_tensor_load_to_lds(g0, g1, z4, z4, 0);
#endif
}
#endif

// ---------------------------------------------------------------------------
// Edge conv: y[o,p] = max_k leaky( s*(W @ [ctr-nbr ; ctr]) + t )
// s > 0  =>  affine+leaky monotone => max taken on raw accumulators.
// Neighbor of point p0+j is (k - p0 - j) mod N  =>  the 51-pt window starts
// at (N - p0 - 31) mod N; LDS row i = point (N-p0-31+i) mod N, and the
// neighbor row for (j,k) is i = k + 31 - j.  Centers are rows 51..82.
// ---------------------------------------------------------------------------
template <int CI, int CO>
__global__ __launch_bounds__(256) void edge_conv_wmma(
    const _Float16* __restrict__ in_act, int in_stride, int in_coff,
    const _Float16* __restrict__ W,   // [CO][2*CI] f16 row-major
    const float* __restrict__ sc, const float* __restrict__ sh,
    _Float16* __restrict__ out_act, int out_stride, int out_coff)
{
  constexpr int WIN  = 51;
  constexpr int ROWS = WIN + 32;
  constexpr int KST  = (2 * CI) / 32;
  constexpr int MT   = CO / 16;
  __shared__ _Float16 sx[ROWS * CI];

  const int b   = blockIdx.y;
  const int p0  = blockIdx.x * 32;
  const int tid = threadIdx.x;
  const _Float16* inb = in_act + (size_t)b * NPTS * in_stride + in_coff;

#if USE_TDM
  if (tid < 32) {  // wave 0 issues the DMAs (TDM ignores EXEC; 1 per WG)
    // window: 51 points starting at (N - p0 - 31) mod N, <=2 wrap segments
    int w0 = NPTS - p0 - 31;
    if (w0 < 0) w0 += NPTS;
    if (w0 >= NPTS) w0 -= NPTS;
    int r1 = NPTS - w0; if (r1 > WIN) r1 = WIN;
    tdm_load_rows(0, inb + (size_t)w0 * in_stride, r1, CI, in_stride);
    if (r1 < WIN)
      tdm_load_rows((unsigned)r1 * CI * 2, inb, WIN - r1, CI, in_stride);
    // centers: 32 points starting at p0, <=2 wrap segments
    int c1 = NPTS - p0; if (c1 > 32) c1 = 32;
    tdm_load_rows((unsigned)WIN * CI * 2, inb + (size_t)p0 * in_stride, c1, CI,
                  in_stride);
    if (c1 < 32)
      tdm_load_rows((unsigned)(WIN + c1) * CI * 2, inb, 32 - c1, CI, in_stride);
    __builtin_amdgcn_s_wait_tensorcnt(0);
  }
#else
  {
    const int per_row = CI / 8;
    uint4* dst = (uint4*)sx;
    for (int t = tid; t < ROWS * per_row; t += 256) {
      int row = t / per_row, cq = t - row * per_row;
      int ap = (row < WIN) ? (NPTS - p0 - 31 + row) : (p0 + (row - WIN));
      if (ap < 0) ap += NPTS;
      if (ap >= NPTS) ap -= NPTS;
      const uint4* src = (const uint4*)(inb + (size_t)ap * in_stride);
      dst[t] = src[cq];
    }
  }
#endif
  __syncthreads();

  const int wave = tid >> 5;     // wave32
  const int lane = tid & 31;
  const int l = lane & 15;
  const int h = lane >> 4;
  const int ssub = wave & 1;     // point subtile
  const int wgrp = wave >> 1;    // M-tile group
  const int j = ssub * 16 + l;   // point within 32-tile
  const int rowc = WIN + j;

  // center chunks: invariant over k and m  (e = [ctr - nbr ; ctr])
  h8 ctr[2 * KST];
  #pragma unroll
  for (int kk = 0; kk < KST; kk++) {
    const int cs = kk * 32 + 16 * h;
    const int co2 = ((CI % 32 == 0) ? (kk * 32 < CI) : (cs < CI)) ? cs : cs - CI;
    const h8* cp = (const h8*)(sx + rowc * CI + co2);
    ctr[2 * kk]     = cp[0];
    ctr[2 * kk + 1] = cp[1];
  }

  for (int m = wgrp; m < MT; m += 4) {
    const int o0 = m * 16;

    // A fragments (weights): reused across all 20 k's
    v16h af[KST];
    const _Float16* wrow = W + (size_t)(o0 + l) * (2 * CI);
    #pragma unroll
    for (int kk = 0; kk < KST; kk++) {
      int c0 = kk * 32;
      h8 alo = *(const h8*)(wrow + c0 + 16 * h);
      h8 ahi = *(const h8*)(wrow + c0 + 16 * h + 8);
      af[kk] = cat16(alo, ahi);
    }

    v8f mx;
    #pragma unroll
    for (int r = 0; r < 8; r++) mx[r] = -3.0e38f;

    for (int k = 0; k < 20; k += 2) {
      const int rn0 = k + 31 - j;     // neighbor rows in window
      const int rn1 = rn0 + 1;
      v8f acc0, acc1;
      #pragma unroll
      for (int r = 0; r < 8; r++) { acc0[r] = 0.0f; acc1[r] = 0.0f; }
      #pragma unroll
      for (int kk = 0; kk < KST; kk++) {
        const int cs = kk * 32 + 16 * h;
        const bool isdiff = (CI % 32 == 0) ? (kk * 32 < CI) : (cs < CI);
        const int cl = isdiff ? cs : 0;
        h8 c0v = ctr[2 * kk], c1v = ctr[2 * kk + 1];
        h8 b00 = c0v, b01 = c1v, b10 = c0v, b11 = c1v;
        if (isdiff) {
          const h8* n0 = (const h8*)(sx + rn0 * CI + cl);
          const h8* n1 = (const h8*)(sx + rn1 * CI + cl);
          b00 = c0v - n0[0]; b01 = c1v - n0[1];
          b10 = c0v - n1[0]; b11 = c1v - n1[1];
        }
        acc0 = __builtin_amdgcn_wmma_f32_16x16x32_f16(
            false, af[kk], false, cat16(b00, b01), (short)0, acc0, false, false);
        acc1 = __builtin_amdgcn_wmma_f32_16x16x32_f16(
            false, af[kk], false, cat16(b10, b11), (short)0, acc1, false, false);
      }
      #pragma unroll
      for (int r = 0; r < 8; r++)
        mx[r] = fmaxf(mx[r], fmaxf(acc0[r], acc1[r]));
    }

    const int p = p0 + j;
    if (p < NPTS) {
      h8 ov;
      #pragma unroll
      for (int r = 0; r < 8; r++) {
        float y = mx[r] * sc[o0 + 8 * h + r] + sh[o0 + 8 * h + r];
        y = (y > 0.0f) ? y : 0.2f * y;
        ov[r] = (_Float16)y;
      }
      *(h8*)(out_act + ((size_t)b * NPTS + p) * out_stride + out_coff + o0 + 8 * h) = ov;
    }
  }
}

// ---------------------------------------------------------------------------
// Pointwise conv GEMM over 32-pt tiles: out = act( s*(W@x) + t )
// Dual even/odd K-chains for WMMA ILP.  OUTH: f16 activations; else f32.
// ---------------------------------------------------------------------------
template <int CIN, int CO, bool LEAKY, bool OUTH>
__global__ __launch_bounds__(256) void gemm_pt_wmma(
    const _Float16* __restrict__ in_act, int in_stride, int in_coff,
    const _Float16* __restrict__ W,   // [CO][CIN] f16
    const float* __restrict__ sc, const float* __restrict__ sh,
    _Float16* __restrict__ out_h, float* __restrict__ out_f,
    int out_stride, int co_real)
{
  constexpr int KST = CIN / 32;      // 12 or 16 (even)
  constexpr int MT  = CO / 16;
  __shared__ _Float16 sx[32 * CIN];

  const int b   = blockIdx.y;
  const int p0  = blockIdx.x * 32;
  const int tid = threadIdx.x;
  const _Float16* inb = in_act + (size_t)b * NPTS * in_stride + in_coff;

#if USE_TDM
  if (tid < 32) {
    int r1 = NPTS - p0; if (r1 > 32) r1 = 32;
    tdm_load_rows(0, inb + (size_t)p0 * in_stride, r1, CIN, in_stride);
    if (r1 < 32)
      tdm_load_rows((unsigned)r1 * CIN * 2, inb, 32 - r1, CIN, in_stride);
    __builtin_amdgcn_s_wait_tensorcnt(0);
  }
#else
  {
    const int per_row = CIN / 8;
    uint4* dst = (uint4*)sx;
    for (int t = tid; t < 32 * per_row; t += 256) {
      int row = t / per_row, cq = t - row * per_row;
      int ap = p0 + row;
      if (ap >= NPTS) ap -= NPTS;
      const uint4* src = (const uint4*)(inb + (size_t)ap * in_stride);
      dst[t] = src[cq];
    }
  }
#endif
  __syncthreads();

  const int wave = tid >> 5, lane = tid & 31;
  const int l = lane & 15, h = lane >> 4;
  const int ssub = wave & 1, wgrp = wave >> 1;
  const int j = ssub * 16 + l;

  for (int m = wgrp; m < MT; m += 4) {
    const int o0 = m * 16;
    v8f acc0, acc1;
    #pragma unroll
    for (int r = 0; r < 8; r++) { acc0[r] = 0.0f; acc1[r] = 0.0f; }

    const _Float16* wrow = W + (size_t)(o0 + l) * CIN;
    #pragma unroll
    for (int kk = 0; kk < KST; kk += 2) {
      const int c0 = kk * 32, c1 = c0 + 32;
      h8 a0l = *(const h8*)(wrow + c0 + 16 * h);
      h8 a0h = *(const h8*)(wrow + c0 + 16 * h + 8);
      h8 a1l = *(const h8*)(wrow + c1 + 16 * h);
      h8 a1h = *(const h8*)(wrow + c1 + 16 * h + 8);
      const h8* b0p = (const h8*)(sx + j * CIN + c0 + 16 * h);
      const h8* b1p = (const h8*)(sx + j * CIN + c1 + 16 * h);
      acc0 = __builtin_amdgcn_wmma_f32_16x16x32_f16(
          false, cat16(a0l, a0h), false, cat16(b0p[0], b0p[1]), (short)0, acc0,
          false, false);
      acc1 = __builtin_amdgcn_wmma_f32_16x16x32_f16(
          false, cat16(a1l, a1h), false, cat16(b1p[0], b1p[1]), (short)0, acc1,
          false, false);
    }

    const int p = p0 + j;
    if (p < NPTS) {
      float y[8];
      #pragma unroll
      for (int r = 0; r < 8; r++) {
        float v = (acc0[r] + acc1[r]) * sc[o0 + 8 * h + r] + sh[o0 + 8 * h + r];
        if (LEAKY) v = (v > 0.0f) ? v : 0.2f * v;
        y[r] = v;
      }
      if (OUTH) {
        h8 ov;
        #pragma unroll
        for (int r = 0; r < 8; r++) ov[r] = (_Float16)y[r];
        *(h8*)(out_h + ((size_t)b * NPTS + p) * out_stride + o0 + 8 * h) = ov;
      } else {
        float* dstp = out_f + ((size_t)b * NPTS + p) * out_stride + o0 + 8 * h;
        if (o0 + 8 * h + 7 < co_real) {      // full slab: 2x b128 stores
          float4 v0, v1;
          v0.x = y[0]; v0.y = y[1]; v0.z = y[2]; v0.w = y[3];
          v1.x = y[4]; v1.y = y[5]; v1.z = y[6]; v1.w = y[7];
          ((float4*)dstp)[0] = v0;
          ((float4*)dstp)[1] = v1;
        } else {
          #pragma unroll
          for (int r = 0; r < 8; r++)
            if (o0 + 8 * h + r < co_real) dstp[r] = y[r];
        }
      }
    }
  }
}

// ---------------------------------------------------------------------------
// Prep kernels
// ---------------------------------------------------------------------------
__global__ __launch_bounds__(256) void prep_x0(
    const float* __restrict__ coords, const float* __restrict__ feats,
    _Float16* __restrict__ x0, float* __restrict__ out_coords,
    float* __restrict__ out_masks)
{
  int i = blockIdx.x * blockDim.x + threadIdx.x;
  if (i >= BATCH * NPTS) return;
  float c0 = coords[3 * i], c1 = coords[3 * i + 1], c2 = coords[3 * i + 2];
  float f3 = feats[4 * i + 3];
  _Float16* d = x0 + (size_t)i * 16;
  d[0] = (_Float16)c0; d[1] = (_Float16)c1; d[2] = (_Float16)c2; d[3] = (_Float16)f3;
  #pragma unroll
  for (int r = 4; r < 16; r++) d[r] = (_Float16)0.0f;
  out_coords[3 * i] = c0; out_coords[3 * i + 1] = c1; out_coords[3 * i + 2] = c2;
  out_masks[i] = 0.0f;
}

// edge weight (co, 2*ci) f32 -> [co][2*cipad] f16 with zero padding per half
__global__ __launch_bounds__(256) void cvt_edge_w(
    const float* __restrict__ src, _Float16* __restrict__ dst,
    int co, int ci, int cipad)
{
  int i = blockIdx.x * blockDim.x + threadIdx.x;
  int total = co * 2 * cipad;
  if (i >= total) return;
  int row = i / (2 * cipad), col = i - row * (2 * cipad);
  float v = 0.0f;
  if (col < cipad) { if (col < ci) v = src[row * 2 * ci + col]; }
  else { int c = col - cipad; if (c < ci) v = src[row * 2 * ci + ci + c]; }
  dst[i] = (_Float16)v;
}

__global__ __launch_bounds__(256) void cvt_plain_w(
    const float* __restrict__ src, _Float16* __restrict__ dst,
    int rows_src, int rows_dst, int cols)
{
  int i = blockIdx.x * blockDim.x + threadIdx.x;
  if (i >= rows_dst * cols) return;
  int row = i / cols;
  dst[i] = (row < rows_src) ? (_Float16)src[i] : (_Float16)0.0f;
}

// fold classifier into f3 weights: Weff[o,c] = sum_m sem_w[o,m]*fs3[m]*fw3[m,c]
__global__ __launch_bounds__(256) void fold_sem_w(
    const float* __restrict__ sem_w, const float* __restrict__ fs3,
    const float* __restrict__ fw3, _Float16* __restrict__ dst)
{
  int i = blockIdx.x * blockDim.x + threadIdx.x;
  if (i >= 32 * 512) return;
  int o = i / 512, c = i - o * 512;
  float acc = 0.0f;
  if (o < 20)
    for (int m = 0; m < 256; m++)
      acc += sem_w[o * 256 + m] * fs3[m] * fw3[m * 512 + c];
  dst[i] = (_Float16)acc;
}

__global__ void fold_sem_b(
    const float* __restrict__ sem_w, const float* __restrict__ ft3,
    const float* __restrict__ sem_b, float* __restrict__ beff,
    float* __restrict__ ones)
{
  int o = threadIdx.x;  // 32 threads
  float acc = 0.0f;
  if (o < 20) {
    acc = sem_b[o];
    for (int m = 0; m < 256; m++) acc += sem_w[o * 256 + m] * ft3[m];
  }
  beff[o] = acc;
  ones[o] = 1.0f;
}

// ---------------------------------------------------------------------------
extern "C" void kernel_launch(void* const* d_in, const int* in_sizes, int n_in,
                              void* d_out, int out_size, void* d_ws, size_t ws_size,
                              hipStream_t stream) {
  const float* coords = (const float*)d_in[0];
  const float* feats  = (const float*)d_in[1];
  const float* ew1 = (const float*)d_in[2],  *es1 = (const float*)d_in[3],  *et1 = (const float*)d_in[4];
  const float* ew2 = (const float*)d_in[5],  *es2 = (const float*)d_in[6],  *et2 = (const float*)d_in[7];
  const float* ew3 = (const float*)d_in[8],  *es3 = (const float*)d_in[9],  *et3 = (const float*)d_in[10];
  const float* ew4 = (const float*)d_in[11], *es4 = (const float*)d_in[12], *et4 = (const float*)d_in[13];
  const float* w5  = (const float*)d_in[14], *s5  = (const float*)d_in[15], *t5  = (const float*)d_in[16];
  const float* fw1 = (const float*)d_in[17], *fs1 = (const float*)d_in[18], *ft1 = (const float*)d_in[19];
  const float* fw2 = (const float*)d_in[20], *fs2 = (const float*)d_in[21], *ft2 = (const float*)d_in[22];
  const float* fw3 = (const float*)d_in[23], *fs3 = (const float*)d_in[24], *ft3 = (const float*)d_in[25];
  const float* semw = (const float*)d_in[26], *semb = (const float*)d_in[27];

  char* ws = (char*)d_ws;
  size_t off = 0;
  auto alloc = [&](size_t bytes) -> void* {
    void* p = ws + off;
    off = (off + bytes + 255) & ~(size_t)255;
    return p;
  };
  const size_t BN = (size_t)BATCH * NPTS;
  _Float16* x0  = (_Float16*)alloc(BN * 16 * 2);
  _Float16* xc  = (_Float16*)alloc(BN * 384 * 2);
  _Float16* yb  = (_Float16*)alloc(BN * 512 * 2);
  _Float16* wE1 = (_Float16*)alloc(64  * 32  * 2);
  _Float16* wE2 = (_Float16*)alloc(64  * 128 * 2);
  _Float16* wE3 = (_Float16*)alloc(128 * 128 * 2);
  _Float16* wE4 = (_Float16*)alloc(128 * 256 * 2);
  _Float16* w5h = (_Float16*)alloc(512 * 384 * 2);
  _Float16* wf1 = (_Float16*)alloc(128 * 512 * 2);
  _Float16* wf2 = (_Float16*)alloc(128 * 512 * 2);
  _Float16* wf3 = (_Float16*)alloc(256 * 512 * 2);
  _Float16* wsm = (_Float16*)alloc(32  * 512 * 2);
  float* semb_eff = (float*)alloc(32 * 4);
  float* ones32   = (float*)alloc(32 * 4);

  float* out  = (float*)d_out;
  float* o_f1 = out;
  float* o_f2 = o_f1 + BN * 128;
  float* o_f3 = o_f2 + BN * 128;
  float* o_co = o_f3 + BN * 256;
  float* o_mk = o_co + BN * 3;
  float* o_sm = o_mk + BN;

  // --- prep ---
  prep_x0<<<(int)((BN + 255) / 256), 256, 0, stream>>>(coords, feats, x0, o_co, o_mk);
  cvt_edge_w<<<(64 * 32   + 255) / 256, 256, 0, stream>>>(ew1, wE1, 64, 4, 16);
  cvt_edge_w<<<(64 * 128  + 255) / 256, 256, 0, stream>>>(ew2, wE2, 64, 64, 64);
  cvt_edge_w<<<(128 * 128 + 255) / 256, 256, 0, stream>>>(ew3, wE3, 128, 64, 64);
  cvt_edge_w<<<(128 * 256 + 255) / 256, 256, 0, stream>>>(ew4, wE4, 128, 128, 128);
  cvt_plain_w<<<(512 * 384 + 255) / 256, 256, 0, stream>>>(w5, w5h, 512, 512, 384);
  cvt_plain_w<<<(128 * 512 + 255) / 256, 256, 0, stream>>>(fw1, wf1, 128, 128, 512);
  cvt_plain_w<<<(128 * 512 + 255) / 256, 256, 0, stream>>>(fw2, wf2, 128, 128, 512);
  cvt_plain_w<<<(256 * 512 + 255) / 256, 256, 0, stream>>>(fw3, wf3, 256, 256, 512);
  fold_sem_w<<<(32 * 512 + 255) / 256, 256, 0, stream>>>(semw, fs3, fw3, wsm);
  fold_sem_b<<<1, 32, 0, stream>>>(semw, ft3, semb, semb_eff, ones32);

  // --- edge convs (x1..x4 land in xc at channel offsets 0/64/128/256) ---
  dim3 gE((NPTS + 31) / 32, BATCH);
  edge_conv_wmma<16, 64>  <<<gE, 256, 0, stream>>>(x0, 16,  0,   wE1, es1, et1, xc, 384, 0);
  edge_conv_wmma<64, 64>  <<<gE, 256, 0, stream>>>(xc, 384, 0,   wE2, es2, et2, xc, 384, 64);
  edge_conv_wmma<64, 128> <<<gE, 256, 0, stream>>>(xc, 384, 64,  wE3, es3, et3, xc, 384, 128);
  edge_conv_wmma<128, 128><<<gE, 256, 0, stream>>>(xc, 384, 128, wE4, es4, et4, xc, 384, 256);

  // --- conv5 (leaky) -> y f16 ---
  gemm_pt_wmma<384, 512, true, true><<<gE, 256, 0, stream>>>(
      xc, 384, 0, w5h, s5, t5, yb, nullptr, 512, 512);

  // --- heads: f32 straight into d_out ---
  gemm_pt_wmma<512, 128, false, false><<<gE, 256, 0, stream>>>(
      yb, 512, 0, wf1, fs1, ft1, nullptr, o_f1, 128, 128);
  gemm_pt_wmma<512, 128, false, false><<<gE, 256, 0, stream>>>(
      yb, 512, 0, wf2, fs2, ft2, nullptr, o_f2, 128, 128);
  gemm_pt_wmma<512, 256, false, false><<<gE, 256, 0, stream>>>(
      yb, 512, 0, wf3, fs3, ft3, nullptr, o_f3, 256, 256);
  // classifier folded through f3: scale=1, shift=b_eff
  gemm_pt_wmma<512, 32, false, false><<<gE, 256, 0, stream>>>(
      yb, 512, 0, wsm, ones32, semb_eff, nullptr, o_sm, 20, 20);
}
